// RGLRU_57123065037324
// MI455X (gfx1250) — matrix operand order repaired and checked by reading, not compiled, not run on hardware
//
#include <hip/hip_runtime.h>

// ---------------------------------------------------------------------------
// RG-LRU for MI455X (gfx1250, wave32, WMMA).
//   gates = x @ W^T  (bf16 WMMA, f32 accum), fused sigmoid/exp/sqrt epilogue,
//   then chunked parallel linear scan  h_t = a_t h_{t-1} + b_t.
// Round 3: fix async-builtin pointer types (AS1/AS3 int4*), keep double-
//          buffered GLOBAL_LOAD_ASYNC_TO_LDS_B128 pipeline.
// ---------------------------------------------------------------------------

#define AS1 __attribute__((address_space(1)))
#define AS3 __attribute__((address_space(3)))

typedef unsigned short u16;
typedef __bf16 bf16_t;
typedef bf16_t v8bf  __attribute__((ext_vector_type(8)));
typedef bf16_t v16bf __attribute__((ext_vector_type(16)));
typedef float  v8f   __attribute__((ext_vector_type(8)));
typedef int    v4i_gnu __attribute__((vector_size(16)));   // builtin's b128 type

#define EMB   2048
#define KDIM  2048
#define SEQ   4096
#define BSZ   4
#define MDIM  (BSZ * SEQ)      // 16384
#define CGATE 8.0f
#define CHUNK 512
#define NCH   (SEQ / CHUNK)    // 8

#define BM  128
#define BN  64                 // per gate half; C tile is BM x (2*BN)
#define BK  32
#define LDK 40                 // padded LDS row stride (halves) to spread banks

// ---------------- f32 -> bf16 (round-to-nearest-even) ----------------------
__device__ __forceinline__ u16 f32_to_bf16(float f) {
  unsigned u = __float_as_uint(f);
  unsigned r = (u + 0x7FFFu + ((u >> 16) & 1u)) >> 16;
  return (u16)r;
}

__device__ __forceinline__ float fast_sigmoid(float v) {
  // v_exp + v_add + v_rcp (avoid IEEE div_scale/div_fixup chains)
  return __builtin_amdgcn_rcpf(1.f + __expf(-v));
}

// GLOBAL_LOAD_ASYNC_TO_LDS_B128: (AS1 int4* src, AS3 int4* dst, imm off, imm cpol)
__device__ __forceinline__ void cp16_async(void* lds, const void* glb) {
  AS1 v4i_gnu* g = (AS1 v4i_gnu*)(unsigned long long)glb;
  AS3 v4i_gnu* l = (AS3 v4i_gnu*)(unsigned)(unsigned long long)lds;
  __builtin_amdgcn_global_load_async_to_lds_b128(g, l, 0, 0);
}
__device__ __forceinline__ void wait_async0() {
  __builtin_amdgcn_s_wait_asynccnt(0);
}

__global__ void cvt_bf16_kernel(const float* __restrict__ src,
                                u16* __restrict__ dst, long n) {
  long i = ((long)blockIdx.x * blockDim.x + threadIdx.x) * 4;
  long stride = (long)gridDim.x * blockDim.x * 4;
  for (; i < n; i += stride) {
    float4 v = *(const float4*)(src + i);
    unsigned lo = (unsigned)f32_to_bf16(v.x) | ((unsigned)f32_to_bf16(v.y) << 16);
    unsigned hi = (unsigned)f32_to_bf16(v.z) | ((unsigned)f32_to_bf16(v.w) << 16);
    uint2 p; p.x = lo; p.y = hi;
    *(uint2*)(dst + i) = p;
  }
}

// ---------------- softplus(a_param) tiled to emb ---------------------------
__global__ void afull_kernel(const float* __restrict__ a_param,
                             float* __restrict__ afull) {
  int i = blockIdx.x * blockDim.x + threadIdx.x;
  if (i < EMB) {
    float v = a_param[i & 127];
    float sp = (v > 0.f) ? (v + log1pf(__expf(-v))) : log1pf(__expf(v));
    afull[i] = sp;
  }
}

// ---------------- GEMM (bf16 WMMA) + fused gate epilogue -------------------
// Block: 256 threads = 8 waves, tile BM x (2*BN) = 128 x 128 of gates.
// Columns [0,64) are gate_x pre-acts (W rows n0..), columns [64,128) are the
// MATCHING gate_a pre-acts (W rows emb+n0..) so pairing happens in registers.
// Double-buffered LDS staged with GLOBAL_LOAD_ASYNC_TO_LDS_B128 (ASYNCcnt);
// one s_wait_asynccnt + one barrier per K-step.
__global__ __launch_bounds__(256)
void gates_gemm_kernel(const u16* __restrict__ Xb,   // [MDIM, KDIM] bf16
                       const u16* __restrict__ Wb,   // [2*EMB, KDIM] bf16
                       const float* __restrict__ x,  // [MDIM, EMB] f32
                       const float* __restrict__ afull, // [EMB]
                       float* __restrict__ Aarr,     // [MDIM, EMB] a-coeffs
                       float* __restrict__ Barr)     // [MDIM, EMB] b-terms
{
  __shared__ u16 As[2][BM][LDK];
  __shared__ u16 Bs[2][2 * BN][LDK];

  const int tid  = threadIdx.x;
  const int lane = tid & 31;
  const int wid  = tid >> 5;   // 0..7
  const int wm   = wid & 1;    // M half (64 rows)
  const int wn   = wid >> 1;   // 0..3 -> 16-col group

  const int m0 = blockIdx.x * BM;
  const int n0 = blockIdx.y * BN;

  v8f zero = {0.f, 0.f, 0.f, 0.f, 0.f, 0.f, 0.f, 0.f};
  v8f acc[4][2];
#pragma unroll
  for (int i = 0; i < 4; i++)
#pragma unroll
    for (int j = 0; j < 2; j++) acc[i][j] = zero;

  // tile-load assignment: thread t copies 16 halves of row (t>>1)
  const int lr = tid >> 1;            // 0..127
  const int lc = (tid & 1) * 16;      // 0 or 16
  const int browg = (lr < BN) ? (n0 + lr) : (EMB + n0 + (lr - BN));

  const int kg = lane >> 4;           // 0 or 1 (K-group within fragment)
  const int lm = lane & 15;

  auto issue_tile = [&](int buf, int k0) {
    const u16* asrc = Xb + (size_t)(m0 + lr) * KDIM + k0 + lc;
    const u16* bsrc = Wb + (size_t)browg * KDIM + k0 + lc;
    cp16_async(&As[buf][lr][lc],     asrc);
    cp16_async(&As[buf][lr][lc + 8], asrc + 8);
    cp16_async(&Bs[buf][lr][lc],     bsrc);
    cp16_async(&Bs[buf][lr][lc + 8], bsrc + 8);
  };

  const int NITER = KDIM / BK;   // 64
  issue_tile(0, 0);

  for (int it = 0; it < NITER; ++it) {
    const int cur = it & 1;
    wait_async0();               // my share of tile `cur` landed in LDS
    __syncthreads();             // everyone's share landed; prev compute done
    if (it + 1 < NITER) issue_tile(cur ^ 1, (it + 1) * BK);

    // A fragments: 16-bit A layout = lane(0..15)=row, K groups {0..7,16..23}
    // for lanes<16 and {8..15,24..31} for lanes>=16.
    v16bf af[4];
#pragma unroll
    for (int fi = 0; fi < 4; fi++) {
      int ra = wm * 64 + fi * 16 + lm;
      v8bf alo = *(const v8bf*)&As[cur][ra][kg * 8];
      v8bf ahi = *(const v8bf*)&As[cur][ra][kg * 8 + 16];
      af[fi] = __builtin_shufflevector(alo, ahi, 0, 1, 2, 3, 4, 5, 6, 7,
                                       8, 9, 10, 11, 12, 13, 14, 15);
    }
    // B fragments: lane(0..15)=col, lanes<16 hold K {0..7,16..23}, lanes>=16 K {8..15,24..31}.
#pragma unroll
    for (int fj = 0; fj < 2; fj++) {
      int cb = fj * 64 + wn * 16 + lm;
      v8bf blo = *(const v8bf*)&Bs[cur][cb][kg * 8];
      v8bf bhi = *(const v8bf*)&Bs[cur][cb][kg * 8 + 16];
      v16bf bfr = __builtin_shufflevector(blo, bhi, 0, 1, 2, 3, 4, 5, 6, 7,
                                          8, 9, 10, 11, 12, 13, 14, 15);
#pragma unroll
      for (int fi = 0; fi < 4; fi++) {
        acc[fi][fj] = __builtin_amdgcn_wmma_f32_16x16x32_bf16(
            false, af[fi], false, bfr, (short)0, acc[fi][fj], false, false);
      }
    }
  }

  // Epilogue: C layout = lanes 0..15 col=lane rows r, lanes 16..31 rows r+8.
  const int e   = n0 + wn * 16 + lm;    // emb channel in [0, 2048)
  const float afe = afull[e];
  const int rhi = (lane >> 4) * 8;
#pragma unroll
  for (int fi = 0; fi < 4; fi++) {
    int mbase = m0 + wm * 64 + fi * 16 + rhi;
#pragma unroll
    for (int r = 0; r < 8; r++) {
      size_t idx = (size_t)(mbase + r) * EMB + e;
      float sx = fast_sigmoid(acc[fi][0][r]);
      float sa = fast_sigmoid(acc[fi][1][r]);
      float a  = __expf(-CGATE * sa * afe);                       // exp(log_a)
      float mult = __builtin_amdgcn_sqrtf(fmaxf(1.f - a * a, 0.f));
      Aarr[idx] = a;
      Barr[idx] = mult * sx * x[idx];
    }
  }
}

// ---------------- chunked linear scan --------------------------------------
// Phase 1: per (batch, chunk, channel): chunk product of a and chunk-local h.
__global__ __launch_bounds__(256)
void scan_carry_kernel(const float* __restrict__ Aarr,
                       const float* __restrict__ Barr,
                       float* __restrict__ cA, float* __restrict__ cH) {
  int e  = blockIdx.x * 256 + threadIdx.x;
  int ch = blockIdx.y;
  int b  = blockIdx.z;
  size_t base = ((size_t)b * SEQ + (size_t)ch * CHUNK) * EMB + e;
  float pa = 1.f, h = 0.f;
  for (int s = 0; s < CHUNK; s++) {
    size_t idx = base + (size_t)s * EMB;
    float a = Aarr[idx];
    float bv = Barr[idx];
    pa *= a;
    h = fmaf(a, h, bv);
  }
  size_t ci = ((size_t)b * NCH + ch) * EMB + e;
  cA[ci] = pa;
  cH[ci] = h;
}

// Phase 2: sequential combine over the NCH chunk carries -> h_in per chunk.
__global__ void scan_combine_kernel(const float* __restrict__ cA,
                                    const float* __restrict__ cH,
                                    float* __restrict__ hin) {
  int t = blockIdx.x * 256 + threadIdx.x;   // 0 .. BSZ*EMB-1
  int b = t / EMB;
  int e = t % EMB;
  float h = 0.f;
  for (int c = 0; c < NCH; c++) {
    size_t ci = ((size_t)b * NCH + c) * EMB + e;
    hin[ci] = h;                            // state entering chunk c
    h = fmaf(cA[ci], h, cH[ci]);
  }
}

// Phase 3: re-scan each chunk from its true h_in; bh holds b on input and h
// on output (in-place, element-wise read-then-write).
__global__ __launch_bounds__(256)
void scan_apply_kernel(const float* __restrict__ Aarr, float* bh,
                       const float* __restrict__ hin) {
  int e  = blockIdx.x * 256 + threadIdx.x;
  int ch = blockIdx.y;
  int b  = blockIdx.z;
  size_t ci = ((size_t)b * NCH + ch) * EMB + e;
  float h = hin[ci];
  size_t base = ((size_t)b * SEQ + (size_t)ch * CHUNK) * EMB + e;
  for (int s = 0; s < CHUNK; s++) {
    size_t idx = base + (size_t)s * EMB;
    float a  = Aarr[idx];
    float bv = bh[idx];
    h = fmaf(a, h, bv);
    bh[idx] = h;
  }
}

// ---------------------------------------------------------------------------
extern "C" void kernel_launch(void* const* d_in, const int* in_sizes, int n_in,
                              void* d_out, int out_size, void* d_ws, size_t ws_size,
                              hipStream_t stream) {
  const float* x  = (const float*)d_in[0];   // [4,4096,2048]
  const float* W  = (const float*)d_in[1];   // [4096,2048]
  const float* ap = (const float*)d_in[2];   // [128]
  float* out = (float*)d_out;                // [4,4096,2048] -> reused as Barr

  const long nX = (long)MDIM * KDIM;         // 33,554,432
  const long nW = (long)2 * EMB * KDIM;      //  8,388,608

  char* ws = (char*)d_ws;
  size_t off = 0;
  auto take = [&](size_t bytes) {
    size_t p = off;
    off += (bytes + 255) & ~(size_t)255;
    return p;
  };
  u16*   Xb    = (u16*)(ws + take((size_t)nX * 2));
  u16*   Wb    = (u16*)(ws + take((size_t)nW * 2));
  float* afull = (float*)(ws + take((size_t)EMB * 4));
  float* Aarr  = (float*)(ws + take((size_t)nX * 4));
  float* cA    = (float*)(ws + take((size_t)BSZ * NCH * EMB * 4));
  float* cH    = (float*)(ws + take((size_t)BSZ * NCH * EMB * 4));
  float* hin   = (float*)(ws + take((size_t)BSZ * NCH * EMB * 4));

  cvt_bf16_kernel<<<8192, 256, 0, stream>>>(x, Xb, nX);
  cvt_bf16_kernel<<<2048, 256, 0, stream>>>(W, Wb, nW);
  afull_kernel<<<(EMB + 255) / 256, 256, 0, stream>>>(ap, afull);

  dim3 ggrid(MDIM / BM, EMB / BN);           // (128, 32)
  gates_gemm_kernel<<<ggrid, 256, 0, stream>>>(Xb, Wb, x, afull, Aarr, out);

  dim3 sgrid(EMB / 256, NCH, BSZ);           // (8, 8, 4)
  scan_carry_kernel<<<sgrid, 256, 0, stream>>>(Aarr, out, cA, cH);
  scan_combine_kernel<<<(BSZ * EMB) / 256, 256, 0, stream>>>(cA, cH, hin);
  scan_apply_kernel<<<sgrid, 256, 0, stream>>>(Aarr, out, hin);
}
